// EfficientTemporalAttention_89361089560601
// MI455X (gfx1250) — matrix operand order repaired and checked by reading, not compiled, not run on hardware
//
#include <hip/hip_runtime.h>
#include <hip/hip_bf16.h>
#include <math.h>

// Problem constants (match reference)
#define SLEN     4096
#define DIM      128
#define HID      64
#define NHEADS   4
#define HDIM     16
#define HALF_WIN 128
#define DECAY_R  0.1f

typedef __attribute__((ext_vector_type(16))) _Float16 v16h;
typedef __attribute__((ext_vector_type(8)))  _Float16 v8h;
typedef __attribute__((ext_vector_type(8)))  float    v8f;

// ---------------------------------------------------------------------------
// WMMA operand layouts for V_WMMA_F32_16X16X32_F16 (wave32, ISA 7.12.2):
//  A (16x32): lane m=lane&15 holds row m;
//     lanes 0-15 : elems[0..7]=K0..7,  elems[8..15]=K16..23
//     lanes 16-31: elems[0..7]=K8..15, elems[8..15]=K24..31
//  B (32x16): lane n=lane&15 holds col n;
//     lanes 0-15 : elems[0..7]=K0..7, elems[8..15]=K16..23 ; lanes 16-31: K8..15, K24..31
//  C/D (16x16 f32, 8 VGPRs): row M = r + 8*(lane>=16), col N = lane&15.
// ---------------------------------------------------------------------------

// A fragment with K valid only in [0,16): lanes 0-15 hold K0..15 (two aligned 16-byte v8h loads), lanes 16-31 zero.
__device__ inline v16h load_a_k16(const _Float16* __restrict__ base, int lane) {
  int m = lane & 15, kh = (lane >> 4) & 1;
  const _Float16* p = base + (size_t)m * HDIM;
  v8h lo = *(const v8h*)p, hi = *(const v8h*)(p + 8);
  v16h a;
#pragma unroll
  for (int i = 0; i < 8; ++i) { a[i] = kh ? (_Float16)0.0f : lo[i]; a[8 + i] = kh ? (_Float16)0.0f : hi[i]; }
  return a;
}

// ---------------------------------------------------------------------------
// Kernel 1: QKV projection. GEMM [B*S,128] x [128,192] -> q/k f16 [B,NH,S,16],
// v f16 transposed [B,NH,16,S]. Wqkv staged in LDS pre-swizzled into
// B-fragment order so inner-loop loads are single contiguous v16h reads.
// 1/sqrt(HD)=0.25 folded into q.
// ---------------------------------------------------------------------------
__global__ void __launch_bounds__(256)
qkv_kernel(const float* __restrict__ seq, const float* __restrict__ Wqkv,
           _Float16* __restrict__ qo, _Float16* __restrict__ ko,
           _Float16* __restrict__ vto) {
  // [kt:4][t:12][lane:32][i:16] f16 = 48KB, each lane's fragment contiguous.
  __shared__ __attribute__((aligned(16))) _Float16 sWf[4 * 12 * 32 * 16];
  int tid = threadIdx.x;
  for (int j = tid; j < DIM * 3 * HID; j += 256) {   // coalesced global read
    int krow = j / (3 * HID);                        // K index 0..127
    int cn   = j % (3 * HID);                        // col 0..191
    int kt = krow >> 5, kk = krow & 31;
    int kh = (kk >> 3) & 1,  ii = (kk & 7) | ((kk >> 4) << 3);
    int t  = cn >> 4,  n  = cn & 15;
    int lane = (kh << 4) | n;
    sWf[(((kt * 12 + t) << 5) | lane) * 16 + ii] = (_Float16)Wqkv[j];
  }
  __syncthreads();

  int wave = tid >> 5, lane = tid & 31;
  int m = lane & 15, kh = (lane >> 4) & 1;
  size_t rowtile = (size_t)blockIdx.x * 128 + (size_t)wave * 16;

  v8f acc[12];
#pragma unroll
  for (int t = 0; t < 12; ++t)
#pragma unroll
    for (int r = 0; r < 8; ++r) acc[t][r] = 0.0f;

#pragma unroll
  for (int kt = 0; kt < 4; ++kt) {
    // A fragment: 4x float4 (128-bit) loads, convert f32->f16.
    const float* row = seq + (rowtile + m) * DIM + kt * 32;
    const float4* rp0 = (const float4*)(row + kh * 8);
    const float4* rp1 = (const float4*)(row + 16 + kh * 8);
    float4 f0 = rp0[0], f1 = rp0[1], f2 = rp1[0], f3 = rp1[1];
    v16h a;
    a[0] = (_Float16)f0.x; a[1] = (_Float16)f0.y; a[2] = (_Float16)f0.z; a[3] = (_Float16)f0.w;
    a[4] = (_Float16)f1.x; a[5] = (_Float16)f1.y; a[6] = (_Float16)f1.z; a[7] = (_Float16)f1.w;
    a[8] = (_Float16)f2.x; a[9] = (_Float16)f2.y; a[10] = (_Float16)f2.z; a[11] = (_Float16)f2.w;
    a[12] = (_Float16)f3.x; a[13] = (_Float16)f3.y; a[14] = (_Float16)f3.z; a[15] = (_Float16)f3.w;

#pragma unroll
    for (int t = 0; t < 12; ++t) {
      v16h b = *(const v16h*)(&sWf[(((kt * 12 + t) << 5) | lane) * 16]);
      acc[t] = __builtin_amdgcn_wmma_f32_16x16x32_f16(
          false, a, false, b, (short)0, acc[t], false, false);
    }
  }

  int n = lane & 15, half = kh;
  int bb = (int)(rowtile >> 12);            // batch (tiles never straddle)
  int s0 = (int)(rowtile & 4095);           // seq pos of tile row 0
#pragma unroll
  for (int t = 0; t < 12; ++t) {
    int g = t >> 2, h = t & 3;
    size_t bh = (size_t)bb * NHEADS + h;
    if (g < 2) {
      _Float16* dst = (g == 0) ? qo : ko;
      float scale = (g == 0) ? 0.25f : 1.0f;       // fold 1/sqrt(16) into q
#pragma unroll
      for (int r = 0; r < 8; ++r)
        dst[(bh * SLEN + s0 + r + half * 8) * HDIM + n] =
            (_Float16)(acc[t][r] * scale);
    } else {
      // v transposed [B,NH,HD,S]: 8 contiguous f16 per lane -> 16B store.
      v8h hv;
#pragma unroll
      for (int r = 0; r < 8; ++r) hv[r] = (_Float16)acc[t][r];
      *(v8h*)(vto + (bh * HDIM + n) * SLEN + s0 + half * 8) = hv;
    }
  }
}

// ---------------------------------------------------------------------------
// Kernel 2: windowed flash attention. One wave per (b, h, 16-query tile).
// Scores and PV via f16 WMMA; online softmax with half-wave shfl reductions.
// Accumulates per-(b, h*16+d) column sums of valid rows into att_sum (atomics).
// ---------------------------------------------------------------------------
__global__ void __launch_bounds__(128)
attn_kernel(const _Float16* __restrict__ q, const _Float16* __restrict__ k,
            const _Float16* __restrict__ vt, const int* __restrict__ lens,
            const float* __restrict__ dw, float* __restrict__ att_sum) {
  __shared__ __attribute__((aligned(16))) _Float16 sP[4][16 * 16];

  int lane = threadIdx.x & 31;
  int wave = threadIdx.x >> 5;
  int widx = blockIdx.x * 4 + wave;     // 0..4095
  int qtile = widx & 255;
  int h     = (widx >> 8) & 3;
  int b     = widx >> 10;
  int qbase = qtile * 16;
  int slen  = lens[b];
  float c   = DECAY_R * dw[0];

  size_t bh = (size_t)b * NHEADS + h;
  const _Float16* qh  = q  + bh * SLEN * HDIM;
  const _Float16* kh_ = k  + bh * SLEN * HDIM;
  const _Float16* vth = vt + bh * HDIM * SLEN;

  int n = lane & 15, half = (lane >> 4) & 1;

  // q fragment (K padded 16->32 with zeros), loaded once: two v8h loads/lane.
  v16h aq = load_a_k16(qh + (size_t)qbase * HDIM, lane);

  // Per-lane row pointers for the contiguous B-fragment loads (half 0 only).
  const _Float16* krow_n = kh_ + (size_t)n * HDIM;        // + kbase*HDIM
  const _Float16* vrow_n = vth + (size_t)n * SLEN;        // + kbase

  v8f o, zc;
  float rmax[8], rsum[8];
#pragma unroll
  for (int r = 0; r < 8; ++r) {
    o[r] = 0.0f; zc[r] = 0.0f; rmax[r] = -1e30f; rsum[r] = 0.0f;
  }

  int kstart = qbase - HALF_WIN; if (kstart < 0) kstart = 0;
  int kend   = qbase + 16 + HALF_WIN; if (kend > SLEN) kend = SLEN;

  for (int kbase = kstart; kbase < kend; kbase += 16) {  // wave-uniform bounds
    // B = k^T fragment: lanes 0-15 read 16 contiguous f16 (row of k); K>=16 -> 0.
    v16h bk;
#pragma unroll
    for (int i = 0; i < 16; ++i) bk[i] = (_Float16)0.0f;
    if (half == 0) bk = *(const v16h*)(krow_n + (size_t)kbase * HDIM);

    v8f s = __builtin_amdgcn_wmma_f32_16x16x32_f16(
        false, aq, false, bk, (short)0, zc, false, false);

    int kj = kbase + n;
    // temporal decay bias: log(exp(-c*(S-1-kj)) + 1e-8)   (per key column)
    float bias = __logf(__expf(-c * (float)(SLEN - 1 - kj)) + 1e-8f);
    bool kok = (kj < slen);

    float pe[8];
#pragma unroll
    for (int r = 0; r < 8; ++r) {
      int qi = qbase + r + half * 8;
      int dlt = qi - kj;
      bool ok = kok && (dlt <= HALF_WIN) && (dlt >= -HALF_WIN);
      float val = ok ? (s[r] + bias) : -1e30f;
      // row max over the 16 lanes holding this row's columns
      float mt = val;
      mt = fmaxf(mt, __shfl_xor(mt, 1, 16));
      mt = fmaxf(mt, __shfl_xor(mt, 2, 16));
      mt = fmaxf(mt, __shfl_xor(mt, 4, 16));
      mt = fmaxf(mt, __shfl_xor(mt, 8, 16));
      float nm = fmaxf(rmax[r], mt);
      float corr = __expf(rmax[r] - nm);        // 1.0 on no-op, 0 after -1e30
      float p = ok ? __expf(val - nm) : 0.0f;
      float ps = p;
      ps += __shfl_xor(ps, 1, 16);
      ps += __shfl_xor(ps, 2, 16);
      ps += __shfl_xor(ps, 4, 16);
      ps += __shfl_xor(ps, 8, 16);
      rsum[r] = rsum[r] * corr + ps;
      rmax[r] = nm;
      o[r] *= corr;
      pe[r] = p;
    }

    // Re-stripe probs (C layout) -> A layout through per-wave LDS slab.
#pragma unroll
    for (int r = 0; r < 8; ++r)
      sP[wave][(r + half * 8) * 16 + n] = (_Float16)pe[r];
    // Same-wave LDS ops are in-order; compiler inserts s_wait_dscnt for RAW.
    v16h ap = load_a_k16(&sP[wave][0], lane);       // two 16-byte LDS loads/lane

    // B = v fragment from transposed v: 16 contiguous f16 per lane; K>=16 -> 0.
    v16h bv;
#pragma unroll
    for (int i = 0; i < 16; ++i) bv[i] = (_Float16)0.0f;
    if (half == 0) bv = *(const v16h*)(vrow_n + kbase);

    o = __builtin_amdgcn_wmma_f32_16x16x32_f16(
        false, ap, false, bv, (short)0, o, false, false);
  }

  // Normalize rows, keep only valid query rows, accumulate column sums.
  float colsum = 0.0f;
#pragma unroll
  for (int r = 0; r < 8; ++r) {
    float val = o[r] / fmaxf(rsum[r], 1e-20f);
    int qi = qbase + r + half * 8;
    if (qi < slen) colsum += val;
  }
  atomicAdd(&att_sum[b * HID + h * HDIM + n], colsum);
}

// ---------------------------------------------------------------------------
// Kernel 0: zero the pooled accumulators (ws is poisoned by the harness).
// ---------------------------------------------------------------------------
__global__ void zero_kernel(float* __restrict__ att_sum,
                            float* __restrict__ seq_sum) {
  int t = threadIdx.x;                 // 256 threads
  att_sum[t] = 0.0f;                   // B*HID = 256
  seq_sum[t] = 0.0f;                   // B*DIM = 512
  seq_sum[t + 256] = 0.0f;
}

// ---------------------------------------------------------------------------
// Kernel 3: masked sum of the raw sequence over valid positions (orig_pooled).
// ---------------------------------------------------------------------------
__global__ void __launch_bounds__(128)
seqsum_kernel(const float* __restrict__ seq, const int* __restrict__ lens,
              float* __restrict__ seq_sum) {
  int b = blockIdx.x >> 5;             // 32 row-chunks per batch
  int chunk = blockIdx.x & 31;
  int d = threadIdx.x;                 // 0..127
  int slen = lens[b];
  int s0 = chunk * 128;
  int s1 = s0 + 128; if (s1 > slen) s1 = slen;
  float acc = 0.0f;
  for (int s = s0; s < s1; ++s)
    acc += seq[((size_t)b * SLEN + s) * DIM + d];
  atomicAdd(&seq_sum[b * DIM + d], acc);
}

// ---------------------------------------------------------------------------
// Kernel 4: output projection of pooled attention + residual pool + layernorm.
// One wave per batch (4 waves total); each lane owns 4 of the 128 channels.
// ---------------------------------------------------------------------------
__global__ void __launch_bounds__(128)
final_kernel(const float* __restrict__ att_sum, const float* __restrict__ seq_sum,
             const float* __restrict__ Wout, const float* __restrict__ b_out,
             const int* __restrict__ lens, const float* __restrict__ gamma,
             const float* __restrict__ beta, float* __restrict__ out) {
  int lane = threadIdx.x & 31;
  int b = threadIdx.x >> 5;            // 0..3
  float denom = fmaxf((float)lens[b], 1.0f);
  float inv_d = 1.0f / denom;

  float x[4];
#pragma unroll
  for (int j = 0; j < 4; ++j) {
    int d = lane + j * 32;
    float acc = 0.0f;
    for (int i = 0; i < HID; ++i)
      acc += att_sum[b * HID + i] * Wout[i * DIM + d];
    // pooled(projected) + b_out + orig_pooled
    x[j] = acc * inv_d + b_out[d] + seq_sum[b * DIM + d] * inv_d;
  }

  float s = x[0] + x[1] + x[2] + x[3];
  for (int off = 1; off < 32; off <<= 1) s += __shfl_xor(s, off, 32);
  float mu = s * (1.0f / DIM);

  float vs = 0.0f;
#pragma unroll
  for (int j = 0; j < 4; ++j) { float dv = x[j] - mu; vs += dv * dv; }
  for (int off = 1; off < 32; off <<= 1) vs += __shfl_xor(vs, off, 32);
  float inv_std = rsqrtf(vs * (1.0f / DIM) + 1e-5f);

#pragma unroll
  for (int j = 0; j < 4; ++j) {
    int d = lane + j * 32;
    out[b * DIM + d] = gamma[d] * (x[j] - mu) * inv_std + beta[d];
  }
}

// ---------------------------------------------------------------------------
// Host launcher. Inputs (reference order): sequence, sequence_lengths, Wqkv,
// Wout, b_out, decay_weight, ln_gamma, ln_beta. Output: [B,128] fp32.
// Workspace: q/k row-major f16 (2MB each), v transposed f16 (2MB),
//            att_sum (1KB), seq_sum (2KB).
// ---------------------------------------------------------------------------
extern "C" void kernel_launch(void* const* d_in, const int* in_sizes, int n_in,
                              void* d_out, int out_size, void* d_ws,
                              size_t ws_size, hipStream_t stream) {
  const float* seq   = (const float*)d_in[0];
  const int*   lens  = (const int*)d_in[1];
  const float* Wqkv  = (const float*)d_in[2];
  const float* Wout  = (const float*)d_in[3];
  const float* b_out = (const float*)d_in[4];
  const float* dw    = (const float*)d_in[5];
  const float* gamma = (const float*)d_in[6];
  const float* beta  = (const float*)d_in[7];
  float* out = (float*)d_out;

  char* ws = (char*)d_ws;
  const size_t qkv_bytes = (size_t)4 * NHEADS * SLEN * HDIM * sizeof(_Float16); // 2MB
  _Float16* q  = (_Float16*)(ws);
  _Float16* k  = (_Float16*)(ws + qkv_bytes);
  _Float16* vt = (_Float16*)(ws + 2 * qkv_bytes);
  float* att_sum = (float*)(ws + 3 * qkv_bytes);       // 4*64 floats
  float* seq_sum = att_sum + 4 * HID;                  // 4*128 floats

  zero_kernel<<<1, 256, 0, stream>>>(att_sum, seq_sum);
  qkv_kernel<<<128, 256, 0, stream>>>(seq, Wqkv, q, k, vt);
  attn_kernel<<<1024, 128, 0, stream>>>(q, k, vt, lens, dw, att_sum);
  seqsum_kernel<<<128, 128, 0, stream>>>(seq, lens, seq_sum);
  final_kernel<<<1, 128, 0, stream>>>(att_sum, seq_sum, Wout, b_out, lens,
                                      gamma, beta, out);
}
